// MIConstrains_20444044329203
// MI455X (gfx1250) — matrix operand order repaired and checked by reading, not compiled, not run on hardware
//
#include <hip/hip_runtime.h>
#include <hip/hip_bf16.h>

// ---------------------------------------------------------------------------
// MI455X (gfx1250, wave32) implementation.
// Stages:
//   1) h = relu(x @ W1 + b1)      f32 WMMA 16x16x4      (3 heads)
//   2) y = h @ W2 + b2            f32 WMMA 16x16x4      (3 heads)
//   3) per-column mean / unbiased std (double accum)
//   4) whiten + row L2-normalize -> f16 embeddings z
//   5) fused logits GEMM + masked logsumexp + positive extraction
//      f16 WMMA 16x16x32, f32 accum, no logits materialization
//   6) deterministic partial reduction -> scalar loss
// ---------------------------------------------------------------------------

typedef _Float16 v16h __attribute__((ext_vector_type(16)));
typedef _Float16 v8h  __attribute__((ext_vector_type(8)));
typedef _Float16 v4h  __attribute__((ext_vector_type(4)));
typedef float    v8f  __attribute__((ext_vector_type(8)));
typedef float    v4f  __attribute__((ext_vector_type(4)));
typedef float    v2f  __attribute__((ext_vector_type(2)));

#define BATCH   8192
#define D_IN    512
#define D_HID   256
#define D_OUT   128

// ---------------------------------------------------------------------------
// Generic f32 WMMA GEMM: O[M x N] = act(X[M x K] @ W[K x N] + bias)
// One wave per 16x16 output tile, V_WMMA_F32_16X16X4_F32 over K.
// A layout (16x4 f32): lane m=l%16, group g=l/16 holds K = {2g, 2g+1}.
// B layout (4x16 f32): lane n=l%16, group g holds rows K = {2g, 2g+1}.
// C layout: VGPR r, lane l -> M = r + 8*(l/16), N = l%16.
// ---------------------------------------------------------------------------
__global__ __launch_bounds__(32)
void gemm_f32_wmma(const float* __restrict__ X, const float* __restrict__ W,
                   const float* __restrict__ bias, float* __restrict__ O,
                   int K, int N, int doRelu)
{
    const int lane = threadIdx.x & 31;
    const int m    = lane & 15;
    const int g    = lane >> 4;
    const int rowBase = blockIdx.x * 16;
    const int colBase = blockIdx.y * 16;

    const float* xrow = X + (size_t)(rowBase + m) * K;
    v8f c = {};
    for (int kb = 0; kb < K; kb += 4) {
        v2f a = *(const v2f*)(xrow + kb + 2 * g);           // contiguous K pair
        const float* wr = W + (size_t)(kb + 2 * g) * N + colBase + m;
        v2f b;
        b.x = wr[0];
        b.y = wr[N];
        c = __builtin_amdgcn_wmma_f32_16x16x4_f32(false, a, false, b,
                                                  (short)0, c, false, false);
    }
    const int col = colBase + m;
    const float bv = bias[col];
#pragma unroll
    for (int r = 0; r < 8; ++r) {
        int row  = rowBase + r + 8 * g;
        float v  = c[r] + bv;
        if (doRelu) v = fmaxf(v, 0.0f);
        O[(size_t)row * N + col] = v;
    }
}

// ---------------------------------------------------------------------------
// Per-column mean + unbiased std (ddof=1) over the batch. One block per
// (head, column); double accumulation for robust statistics.
// ---------------------------------------------------------------------------
__global__ __launch_bounds__(256)
void colstats(const float* __restrict__ Y, float* __restrict__ mu,
              float* __restrict__ rsd)
{
    const int col  = blockIdx.x & (D_OUT - 1);
    const int head = blockIdx.x >> 7;
    const float* yp = Y + (size_t)head * BATCH * D_OUT + col;

    double s = 0.0, s2 = 0.0;
    for (int r = threadIdx.x; r < BATCH; r += 256) {
        float v = yp[(size_t)r * D_OUT];
        s  += (double)v;
        s2 += (double)v * (double)v;
    }
    __shared__ double sh[256], sh2[256];
    sh[threadIdx.x]  = s;
    sh2[threadIdx.x] = s2;
    __syncthreads();
    for (int o = 128; o > 0; o >>= 1) {
        if (threadIdx.x < o) {
            sh[threadIdx.x]  += sh[threadIdx.x + o];
            sh2[threadIdx.x] += sh2[threadIdx.x + o];
        }
        __syncthreads();
    }
    if (threadIdx.x == 0) {
        double mean = sh[0] / (double)BATCH;
        double var  = (sh2[0] - sh[0] * sh[0] / (double)BATCH) / (double)(BATCH - 1);
        if (var < 0.0) var = 0.0;
        double sd = sqrt(var) + 1e-5;                // torch-style: std + eps
        mu[blockIdx.x]  = (float)mean;
        rsd[blockIdx.x] = (float)(1.0 / sd);
    }
}

// ---------------------------------------------------------------------------
// Whiten each column, then L2-normalize each row, emit f16 embeddings.
// One wave per row (128 cols = 4 per lane).
// ---------------------------------------------------------------------------
__global__ __launch_bounds__(256)
void whiten_norm(const float* __restrict__ Y, const float* __restrict__ mu,
                 const float* __restrict__ rsd, _Float16* __restrict__ Z)
{
    const int wave   = threadIdx.x >> 5;
    const int lane   = threadIdx.x & 31;
    const int rowIdx = blockIdx.x * 8 + wave;        // 0 .. 3*BATCH-1
    const int head   = rowIdx >> 13;
    const int c0     = lane * 4;

    const float* yr = Y + (size_t)rowIdx * D_OUT;
    v4f v = *(const v4f*)(yr + c0);
    float w[4];
    float ss = 0.0f;
#pragma unroll
    for (int i = 0; i < 4; ++i) {
        int sc = head * D_OUT + c0 + i;
        w[i] = (v[i] - mu[sc]) * rsd[sc];
        ss += w[i] * w[i];
    }
#pragma unroll
    for (int o = 16; o > 0; o >>= 1) ss += __shfl_xor(ss, o, 32);
    float nrm = fmaxf(sqrtf(ss), 1e-12f);
    float inv = 1.0f / nrm;

    v4h z;
#pragma unroll
    for (int i = 0; i < 4; ++i) z[i] = (_Float16)(w[i] * inv);
    *(v4h*)(Z + (size_t)rowIdx * D_OUT + c0) = z;
}

// ---------------------------------------------------------------------------
// Fused pair loss. Pair p picks heads (h1,h2) in combinations order:
// (B,E),(B,F),(E,F). Each 1-wave block owns 16 rows of v1, streams all
// 16384 columns of cat=[v1;v2] in 16-wide tiles, 4x v_wmma_f32_16x16x32_f16
// per tile, and folds tiles into running exp-sums. Diagonal of the first
// block masked to logit 0 (exp contributes 1); positive logit (j == b+i)
// captured from the second block. Logits are bounded by 1, so the unshifted
// logsumexp is numerically safe.
//
// f16 A 16x32: lane m=l%16, group g holds K = {8g..8g+7, 16+8g..16+8g+7}.
// f16 B 32x16: lane n=l%16, group g holds K = {8g..8g+7, 16+8g..16+8g+7}.
// ---------------------------------------------------------------------------
__global__ __launch_bounds__(32)
void pair_loss(const _Float16* __restrict__ Z, float* __restrict__ partials)
{
    const int p  = blockIdx.y;                 // 0..2
    const int h1 = (p == 2) ? 1 : 0;
    const int h2 = (p == 0) ? 1 : 2;
    const _Float16* z1 = Z + (size_t)h1 * BATCH * D_OUT;
    const _Float16* z2 = Z + (size_t)h2 * BATCH * D_OUT;

    const int lane = threadIdx.x & 31;
    const int n    = lane & 15;
    const int g    = lane >> 4;
    const int rowBase = blockIdx.x * 16;

    // Preload the 4 A fragments (full K=128) for this wave's 16 rows.
    const _Float16* arow = z1 + (size_t)(rowBase + n) * D_OUT;
    v16h afrag[4];
#pragma unroll
    for (int kb = 0; kb < 4; ++kb) {
        v8h lo = *(const v8h*)(arow + kb * 32 + g * 8);
        v8h hi = *(const v8h*)(arow + kb * 32 + 16 + g * 8);
        afrag[kb] = __builtin_shufflevector(lo, hi, 0, 1, 2, 3, 4, 5, 6, 7,
                                            8, 9, 10, 11, 12, 13, 14, 15);
    }

    float psum[8] = {0, 0, 0, 0, 0, 0, 0, 0};
    float pos[8]  = {0, 0, 0, 0, 0, 0, 0, 0};

    for (int ct = 0; ct < 2 * BATCH / 16; ++ct) {
        const int colBase = ct * 16;
        const bool first  = colBase < BATCH;
        const _Float16* src = first ? z1 : z2;
        const int j0 = first ? colBase : (colBase - BATCH);
        const _Float16* brow = src + (size_t)(j0 + n) * D_OUT;

        v8f c = {};
#pragma unroll
        for (int kb = 0; kb < 4; ++kb) {
            v8h lo = *(const v8h*)(brow + kb * 32 + g * 8);
            v8h hi = *(const v8h*)(brow + kb * 32 + 16 + g * 8);
            v16h bfrag = __builtin_shufflevector(lo, hi, 0, 1, 2, 3, 4, 5, 6, 7,
                                                 8, 9, 10, 11, 12, 13, 14, 15);
            c = __builtin_amdgcn_wmma_f32_16x16x32_f16(false, afrag[kb], false,
                                                       bfrag, (short)0, c,
                                                       false, false);
        }

        const int jn = j0 + n;                 // row index inside src block
#pragma unroll
        for (int r = 0; r < 8; ++r) {
            const int i = rowBase + r + 8 * g; // global row of this element
            float logit = c[r];
            if (first) {
                if (jn == i) logit = 0.0f;     // mask self-similarity
            } else {
                if (jn == i) pos[r] = logit;   // positive logit
            }
            psum[r] += __expf(logit);
        }
    }

    // Reduce across the 16 lanes of each half-wave (rows r / r+8), then
    // combine halves. Lane 0 writes one deterministic partial per block.
    float partial = 0.0f;
#pragma unroll
    for (int r = 0; r < 8; ++r) {
        float s = psum[r];
        float q = pos[r];
#pragma unroll
        for (int o = 1; o <= 8; o <<= 1) {
            s += __shfl_xor(s, o, 32);
            q += __shfl_xor(q, o, 32);
        }
        if (n == 0) partial += __logf(s) - q;  // -log p(positive) for one row
    }
    partial += __shfl_xor(partial, 16, 32);
    if (lane == 0) partials[p * (BATCH / 16) + blockIdx.x] = partial;
}

__global__ __launch_bounds__(256)
void finalize(const float* __restrict__ partials, float* __restrict__ out)
{
    __shared__ float sh[256];
    float s = 0.0f;
    for (int i = threadIdx.x; i < 3 * (BATCH / 16); i += 256) s += partials[i];
    sh[threadIdx.x] = s;
    __syncthreads();
    for (int o = 128; o > 0; o >>= 1) {
        if (threadIdx.x < o) sh[threadIdx.x] += sh[threadIdx.x + o];
        __syncthreads();
    }
    if (threadIdx.x == 0) out[0] = sh[0] / (3.0f * (float)BATCH);
}

// ---------------------------------------------------------------------------
extern "C" void kernel_launch(void* const* d_in, const int* in_sizes, int n_in,
                              void* d_out, int out_size, void* d_ws, size_t ws_size,
                              hipStream_t stream)
{
    (void)in_sizes; (void)n_in; (void)out_size; (void)ws_size;

    const float* x[3]  = {(const float*)d_in[0], (const float*)d_in[1],  (const float*)d_in[2]};
    const float* W1[3] = {(const float*)d_in[3], (const float*)d_in[7],  (const float*)d_in[11]};
    const float* b1[3] = {(const float*)d_in[4], (const float*)d_in[8],  (const float*)d_in[12]};
    const float* W2[3] = {(const float*)d_in[5], (const float*)d_in[9],  (const float*)d_in[13]};
    const float* b2[3] = {(const float*)d_in[6], (const float*)d_in[10], (const float*)d_in[14]};

    // Workspace layout (floats). zh reuses the dead `h` region.
    float* ws_f = (float*)d_ws;
    float* h    = ws_f;                                   // 3*8192*256
    float* y    = ws_f + (size_t)3 * BATCH * D_HID;       // 3*8192*128
    float* mu   = y + (size_t)3 * BATCH * D_OUT;          // 384
    float* rsd  = mu + 3 * D_OUT;                         // 384
    _Float16* zh = (_Float16*)ws_f;                       // 3*8192*128 f16 (over h)
    float* partials = ws_f + ((size_t)3 * BATCH * D_OUT) / 2;  // after zh, inside h

    for (int m = 0; m < 3; ++m)
        gemm_f32_wmma<<<dim3(BATCH / 16, D_HID / 16), 32, 0, stream>>>(
            x[m], W1[m], b1[m], h + (size_t)m * BATCH * D_HID, D_IN, D_HID, 1);

    for (int m = 0; m < 3; ++m)
        gemm_f32_wmma<<<dim3(BATCH / 16, D_OUT / 16), 32, 0, stream>>>(
            h + (size_t)m * BATCH * D_HID, W2[m], b2[m],
            y + (size_t)m * BATCH * D_OUT, D_HID, D_OUT, 0);

    colstats<<<3 * D_OUT, 256, 0, stream>>>(y, mu, rsd);
    whiten_norm<<<3 * BATCH / 8, 256, 0, stream>>>(y, mu, rsd, zh);
    pair_loss<<<dim3(BATCH / 16, 3), 32, 0, stream>>>(zh, partials);
    finalize<<<1, 256, 0, stream>>>(partials, (float*)d_out);
}
